// DyRepNode_37203006718619
// MI455X (gfx1250) — matrix operand order repaired and checked by reading, not compiled, not run on hardware
//
#include <hip/hip_runtime.h>

// DyRep-style temporal-graph scan for MI455X (gfx1250, wave32, WMMA).
// One persistent workgroup (8 waves) walks the sequential 512-step scan.
// The four 128x128 weight matrices live in VGPRs as f16 WMMA B-fragments
// (loaded once). Each step: two K-concatenated f16 WMMA GEMMs (24 v_wmma),
// vectorized LDS fragment assembly (ds_load_b128 + shufflevector, no
// branchy per-element selects), L2-resident gather/scatter of z rows.

#define NNODES 100000
#define HDIM   128
#define BEV    512
#define DNB    32
#define NEGS   5

typedef __attribute__((ext_vector_type(16))) _Float16 v16h;
typedef __attribute__((ext_vector_type(8)))  _Float16 v8h;
typedef __attribute__((ext_vector_type(4)))  _Float16 v4h;
typedef __attribute__((ext_vector_type(8)))  float    v8f;

#define CAT16(lo, hi) __builtin_shufflevector(lo, hi, 0,1,2,3,4,5,6,7,8,9,10,11,12,13,14,15)

__device__ __forceinline__ float sigmoidf_(float x)  { return 1.0f / (1.0f + __expf(-x)); }
__device__ __forceinline__ float softplusf_(float x) { return (x > 20.0f) ? x : log1pf(__expf(x)); }

__launch_bounds__(256, 1)
__global__ void dyrep_scan_kernel(
    const float* __restrict__ time_bar,  const float* __restrict__ time_cur,
    const float* __restrict__ time_delta,const int*   __restrict__ u_arr,
    const int*   __restrict__ nb_arr,    const int*   __restrict__ neg_arr,
    const float* __restrict__ W_omega,   const float* __restrict__ b_omega,
    const float* __restrict__ W_h,       const float* __restrict__ b_h,
    const float* __restrict__ W_e2n,     const float* __restrict__ b_e2n,
    const float* __restrict__ W_rec_e,   const float* __restrict__ b_rec_e,
    const float* __restrict__ W_rec_n,   const float* __restrict__ b_rec_n,
    const float* __restrict__ W_time,    const float* __restrict__ b_time,
    const float* __restrict__ w_t_p,     const float* __restrict__ alpha_p,
    const float* __restrict__ psi_p,
    float* __restrict__ z, float* __restrict__ out)
{
  __shared__ __align__(16) _Float16 sA[DNB][HDIM];   // z_nb f16 (GEMM1 A, K-low)
  __shared__ __align__(16) _Float16 sZu16[HDIM];     // z_u f16 (K-high, row-uniform)
  __shared__ __align__(16) _Float16 sM16[HDIM];      // mean(z_nb) f16 (GEMM2 row 0)
  __shared__ __align__(16) float sH[DNB][HDIM];      // h_nb staging (coalesced WB)
  __shared__ __align__(16) float sZu[HDIM];          // z_u f32 (Hawkes)
  __shared__ float sWom[HDIM];
  __shared__ float sWt[HDIM];
  __shared__ float sBiasU[HDIM];                     // b_h + b_rec_e + b_time
  __shared__ float sBiasNb[HDIM];                    // b_e2n + b_rec_n + b_time
  __shared__ float sTd1[DNB];
  __shared__ int   sNb[DNB];
  __shared__ int   sNeg[NEGS];
  __shared__ float sTd0;

  const int tid  = threadIdx.x;
  const int wave = tid >> 5;
  const int lane = tid & 31;

  if (tid < HDIM) {
    sWom[tid]    = W_omega[tid];
    sWt[tid]     = W_time[tid];
    sBiasU[tid]  = b_h[tid] + b_rec_e[tid] + b_time[tid];
    sBiasNb[tid] = b_e2n[tid] + b_rec_n[tid] + b_time[tid];
  }

  const float b_om = b_omega[0];
  const float wts  = w_t_p[0];
  const float alp  = alpha_p[0];
  const float psi  = psi_p[0];
  const float inv_psi = 1.0f / psi;

  // ---- Persistent B fragments (loaded once). Wave w owns cols [16w, 16w+16).
  // B layout (16x16x32 f16): lane<16 -> col n0+lane, K=k0+{0..7,16..23};
  //                          lane>=16 -> col n0+lane-16, K=k0+{8..15,24..31}.
  const int n0    = wave * 16;
  const int col   = n0 + (lane & 15);
  const int kHalf = (lane >> 4) * 8;
  v16h b1[8], b2[8];   // B1=[W_rec_n;W_e2n], B2=[W_h;W_rec_e] along K=256
  #pragma unroll
  for (int kk = 0; kk < 8; ++kk) {
    #pragma unroll
    for (int e = 0; e < 16; ++e) {
      const int kg = kk * 32 + kHalf + (e & 7) + (e >> 3) * 16;
      float w1, w2;
      if (kg < 128) { w1 = W_rec_n[kg * HDIM + col];        w2 = W_h[kg * HDIM + col]; }
      else          { w1 = W_e2n[(kg - 128) * HDIM + col];  w2 = W_rec_e[(kg - 128) * HDIM + col]; }
      b1[kk][e] = (_Float16)w1;
      b2[kk][e] = (_Float16)w2;
    }
  }
  __syncthreads();

  for (int it = 0; it < BEV; ++it) {
    const float* tbrow = time_bar + (size_t)it * NNODES;
    const float  tc    = time_cur[it];
    const int    u     = u_arr[it];

    // ---- Phase A1: step scalars / indices
    if (tid < DNB) {
      sNb[tid]  = nb_arr[it * DNB + tid];
      sTd1[tid] = tc - tbrow[1 + tid];        // positional, as in reference
    }
    if (tid < NEGS) sNeg[tid] = neg_arr[it * NEGS + tid];
    if (tid == 0)   sTd0 = tc - tbrow[0];
    __syncthreads();

    // ---- Phase A2: gather pre-update rows (b128 loads, packed f16 stores)
    if (tid < 32) {
      const float4 vv = *(const float4*)(z + (size_t)u * HDIM + tid * 4);
      *(float4*)(&sZu[tid * 4]) = vv;
      const v4h hv = { (_Float16)vv.x, (_Float16)vv.y, (_Float16)vv.z, (_Float16)vv.w };
      *(v4h*)(&sZu16[tid * 4]) = hv;
    }
    #pragma unroll
    for (int q = tid; q < DNB * 32; q += 256) {   // 32 rows x 32 float4
      const int r = q >> 5, c4 = (q & 31) * 4;
      const float4 vv = *(const float4*)(z + (size_t)sNb[r] * HDIM + c4);
      const v4h hv = { (_Float16)vv.x, (_Float16)vv.y, (_Float16)vv.z, (_Float16)vv.w };
      *(v4h*)(&sA[r][c4]) = hv;
    }
    __syncthreads();

    // ---- Phase B: Hawkes (waves 0..5) + neighbor mean (wave 6, b128 reads)
    if (wave < 6) {
      float part = 0.0f;
      if (wave == 0) {
        #pragma unroll
        for (int k = 0; k < 4; ++k) part += sZu[lane + 32 * k] * sWom[lane + 32 * k];
      } else {
        const float* zr = z + (size_t)sNeg[wave - 1] * HDIM;
        #pragma unroll
        for (int k = 0; k < 4; ++k) part += zr[lane + 32 * k] * sWom[lane + 32 * k];
      }
      part += __shfl_down(part, 16, 32);
      part += __shfl_down(part, 8, 32);
      part += __shfl_down(part, 4, 32);
      part += __shfl_down(part, 2, 32);
      part += __shfl_down(part, 1, 32);
      if (lane == 0) {
        const float td  = (wave == 0) ? (tc - time_delta[it * 2]) : (tc - tbrow[sNeg[wave - 1]]);
        const float g   = part + b_om + alp * __expf(-wts * td);
        const float lam = psi * softplusf_(g * inv_psi);
        if (wave == 0) out[it] = lam;
        else           out[BEV + it * NEGS + (wave - 1)] = lam * (1.0f / NEGS);
      }
    } else if (wave == 6 && lane < 16) {
      float s[8] = {0,0,0,0,0,0,0,0};
      #pragma unroll
      for (int r = 0; r < DNB; ++r) {
        const v8h p = *(const v8h*)(&sA[r][lane * 8]);
        #pragma unroll
        for (int j = 0; j < 8; ++j) s[j] += (float)p[j];
      }
      v8h m;
      #pragma unroll
      for (int j = 0; j < 8; ++j) m[j] = (_Float16)(s[j] * (1.0f / 32.0f));
      *(v8h*)(&sM16[lane * 8]) = m;
    }
    __syncthreads();

    // ---- Phase C: fused WMMA GEMMs over concatenated K=256
    // GEMM1: [z_nb | z_u] @ [W_rec_n ; W_e2n] -> 32x128 h_nb preactivation
    // GEMM2: [mean | z_u] @ [W_h ; W_rec_e]   -> row 0 = h_u preactivation
    v8f acc1a = {}; v8f acc1b = {}; v8f acc2 = {};
    const int  rlo    = lane & 15;
    const int  kh8    = (lane >> 4) * 8;       // A layout: two 8-half chunks
    const bool isrow0 = (rlo == 0);
    const v16h z16    = {};
    const _Float16* rowA = &sA[rlo][0];
    const _Float16* rowB = &sA[16 + rlo][0];
    #pragma unroll
    for (int kk = 0; kk < 8; ++kk) {
      v16h a1a, a1b, a2;
      if (kk < 4) {                            // K-low: per-row data from LDS
        const int base = kk * 32 + kh8;
        v8h lo = *(const v8h*)(rowA + base);
        v8h hi = *(const v8h*)(rowA + base + 16);
        a1a = CAT16(lo, hi);
        lo = *(const v8h*)(rowB + base);
        hi = *(const v8h*)(rowB + base + 16);
        a1b = CAT16(lo, hi);
        lo = *(const v8h*)(&sM16[base]);       // unconditional broadcast load
        hi = *(const v8h*)(&sM16[base + 16]);
        const v16h am = CAT16(lo, hi);
        a2 = isrow0 ? am : z16;                // vector cndmask, no branches
      } else {                                 // K-high: z_u, row-uniform
        const int base = (kk - 4) * 32 + kh8;
        const v8h lo = *(const v8h*)(&sZu16[base]);
        const v8h hi = *(const v8h*)(&sZu16[base + 16]);
        const v16h az = CAT16(lo, hi);
        a1a = az; a1b = az;
        a2  = isrow0 ? az : z16;
      }
      acc1a = __builtin_amdgcn_wmma_f32_16x16x32_f16(false, a1a, false, b1[kk], (short)0, acc1a, false, false);
      acc1b = __builtin_amdgcn_wmma_f32_16x16x32_f16(false, a1b, false, b1[kk], (short)0, acc1b, false, false);
      acc2  = __builtin_amdgcn_wmma_f32_16x16x32_f16(false, a2,  false, b2[kk], (short)0, acc2,  false, false);
    }

    // ---- Phase D: z[u] = h_u (row 0 = acc2[0], lanes 0..15) + stage h_nb
    if (lane < 16) {
      const int n = n0 + lane;
      z[(size_t)u * HDIM + n] = sigmoidf_(acc2[0] + sBiasU[n] + sTd0 * sWt[n]);
    }
    {
      const int   n     = n0 + rlo;
      const int   mbase = (lane < 16) ? 0 : 8;  // C/D layout
      const float wtn   = sWt[n], bnb = sBiasNb[n];
      #pragma unroll
      for (int e = 0; e < 8; ++e) {
        const int m1 = mbase + e;
        sH[m1][n]      = sigmoidf_(acc1a[e] + bnb + sTd1[m1] * wtn);
        const int m2 = 16 + mbase + e;
        sH[m2][n]      = sigmoidf_(acc1b[e] + bnb + sTd1[m2] * wtn);
      }
    }
    asm volatile("s_wait_storecnt 0x0" ::: "memory");
    __syncthreads();   // h_u committed before h_nb (JAX scatter order: nb wins)

    // ---- Phase E: coalesced h_nb writeback (b128 stores per row)
    #pragma unroll
    for (int q = tid; q < DNB * 32; q += 256) {
      const int r = q >> 5, c4 = (q & 31) * 4;
      *(float4*)(z + (size_t)sNb[r] * HDIM + c4) = *(const float4*)(&sH[r][c4]);
    }

    // Warm next step's gather rows in cache (global_prefetch_b8, hint only)
    if (it + 1 < BEV) {
      if (tid < DNB)       __builtin_prefetch(z + (size_t)nb_arr[(it + 1) * DNB + tid] * HDIM, 0, 1);
      else if (tid == DNB) __builtin_prefetch(z + (size_t)u_arr[it + 1] * HDIM, 0, 1);
    }
    asm volatile("s_wait_storecnt 0x0" ::: "memory");
    __syncthreads();   // stores visible before next step's gathers
  }
}

extern "C" void kernel_launch(void* const* d_in, const int* in_sizes, int n_in,
                              void* d_out, int out_size, void* d_ws, size_t ws_size,
                              hipStream_t stream) {
  (void)in_sizes; (void)n_in; (void)out_size; (void)ws_size;
  const float* time_bar   = (const float*)d_in[0];
  const float* time_cur   = (const float*)d_in[1];
  const float* time_delta = (const float*)d_in[2];
  const int*   u_arr      = (const int*)d_in[3];
  const int*   nb_arr     = (const int*)d_in[4];
  const int*   neg_arr    = (const int*)d_in[5];
  const float* z0         = (const float*)d_in[6];
  const float* W_omega    = (const float*)d_in[7];
  const float* b_omega    = (const float*)d_in[8];
  const float* W_h        = (const float*)d_in[9];
  const float* b_h        = (const float*)d_in[10];
  const float* W_e2n      = (const float*)d_in[11];
  const float* b_e2n      = (const float*)d_in[12];
  const float* W_rec_e    = (const float*)d_in[13];
  const float* b_rec_e    = (const float*)d_in[14];
  const float* W_rec_n    = (const float*)d_in[15];
  const float* b_rec_n    = (const float*)d_in[16];
  const float* W_time     = (const float*)d_in[17];
  const float* b_time     = (const float*)d_in[18];
  const float* w_t        = (const float*)d_in[19];
  const float* alpha      = (const float*)d_in[20];
  const float* psi        = (const float*)d_in[21];

  float* z = (float*)d_ws;  // mutable copy of z0 (inputs must not be mutated)
  hipMemcpyAsync(z, z0, (size_t)NNODES * HDIM * sizeof(float),
                 hipMemcpyDeviceToDevice, stream);

  dyrep_scan_kernel<<<dim3(1), dim3(256), 0, stream>>>(
      time_bar, time_cur, time_delta, u_arr, nb_arr, neg_arr,
      W_omega, b_omega, W_h, b_h, W_e2n, b_e2n, W_rec_e, b_rec_e,
      W_rec_n, b_rec_n, W_time, b_time, w_t, alpha, psi,
      z, (float*)d_out);
}
